// GCNEncoder_68324339745256
// MI455X (gfx1250) — matrix operand order repaired and checked by reading, compiled, not run on hardware
//
#include <hip/hip_runtime.h>
#include <hip/hip_bf16.h>

typedef __bf16 bf16;
typedef __attribute__((ext_vector_type(16))) __bf16 v16bf;
typedef __attribute__((ext_vector_type(8)))  float  v8f;

#define N_NODES 10000
#define N_EDGES 150000
#define N_GRAPHS 64

// ---------------- utility kernels ----------------

__global__ void zero_i32(int* p, int n) {
  int i = blockIdx.x * blockDim.x + threadIdx.x;
  if (i < n) p[i] = 0;
}
__global__ void zero_f32(float* p, int n) {
  int i = blockIdx.x * blockDim.x + threadIdx.x;
  if (i < n) p[i] = 0.0f;
}
__global__ void copy_i32(const int* __restrict__ s, int* __restrict__ d, int n) {
  int i = blockIdx.x * blockDim.x + threadIdx.x;
  if (i < n) d[i] = s[i];
}
__global__ void cvt_f32_bf16(const float* __restrict__ in, bf16* __restrict__ out, int n) {
  int i = blockIdx.x * blockDim.x + threadIdx.x;
  if (i < n) out[i] = (bf16)in[i];
}
// W: [fi][fo] fp32 row-major -> Wt: [fo][fi] bf16 row-major
__global__ void transpose_w(const float* __restrict__ W, bf16* __restrict__ Wt, int fi, int fo) {
  int i = blockIdx.x * blockDim.x + threadIdx.x;
  if (i >= fi * fo) return;
  int r = i / fo, c = i % fo;
  Wt[(size_t)c * fi + r] = (bf16)W[i];
}

// ---------------- graph preprocessing ----------------

__global__ void count_deg(const int* __restrict__ dst, int* counts, int E) {
  int e = blockIdx.x * blockDim.x + threadIdx.x;
  if (e < E) atomicAdd(&counts[dst[e]], 1);
}
__global__ void make_dinv(const int* __restrict__ counts, float* __restrict__ dinv, int n) {
  int i = blockIdx.x * blockDim.x + threadIdx.x;
  if (i < n) dinv[i] = rsqrtf((float)counts[i] + 1.0f);  // +1 self loop; always > 0
}
// single-block exclusive scan of counts -> row_ptr[0..n]
__global__ __launch_bounds__(1024)
void scan_rowptr(const int* __restrict__ counts, int* __restrict__ row_ptr, int n) {
  __shared__ int buf[1024];
  __shared__ int carry;
  int tid = threadIdx.x;
  if (tid == 0) carry = 0;
  __syncthreads();
  for (int base = 0; base < n; base += 1024) {
    int i = base + tid;
    int v = (i < n) ? counts[i] : 0;
    buf[tid] = v;
    __syncthreads();
    for (int off = 1; off < 1024; off <<= 1) {
      int t = (tid >= off) ? buf[tid - off] : 0;
      __syncthreads();
      buf[tid] += t;
      __syncthreads();
    }
    int inc = buf[tid];
    if (i < n) row_ptr[i] = carry + inc - v;  // exclusive
    __syncthreads();
    if (tid == 1023) carry += buf[1023];
    __syncthreads();
  }
  if (tid == 0) row_ptr[n] = carry;
}
__global__ void fill_edges(const int* __restrict__ src, const int* __restrict__ dst,
                           int* cursor, int* __restrict__ esrc, int E) {
  int e = blockIdx.x * blockDim.x + threadIdx.x;
  if (e >= E) return;
  int d = dst[e];
  int p = atomicAdd(&cursor[d], 1);
  esrc[p] = src[e];
}

// ---------------- WMMA GEMM: C[M,N] = A[M,K](bf16) * Wt[N,K]^T (bf16), fp32 out ----------------
// Double-buffered LDS, register-prefetched b128 tile loads, clamped (branch-free) edge handling.

#define BM 128
#define BN 64
#define BK 32
#define ASTR 48   // padded LDS row stride (elements); 96B = 16B-aligned for b128
#define BSTR 48

__global__ __launch_bounds__(256)
void gemm_bf16_wmma(const bf16* __restrict__ A, const bf16* __restrict__ Wt,
                    float* __restrict__ C, int M, int K, int N) {
  __shared__ bf16 sA[2 * BM * ASTR];
  __shared__ bf16 sB[2 * BN * BSTR];
  const int tid  = threadIdx.x;
  const int lane = tid & 31;
  const int wave = tid >> 5;           // 8 waves: one 16-row strip each
  const int rowBase = blockIdx.x * BM;
  const int colBase = blockIdx.y * BN;

  v8f acc[4] = {};                      // 16x64 per wave = four 16x16 tiles

  // register prefetch staging (16B chunks of bf16 data)
  float4 pa0, pa1, pb;

  auto gload = [&](int k0) {
    // A tile: 128 rows x 64B -> 512 b128 loads -> 2 per thread (row clamped, no divergence)
    {
      int idx = tid;
      int r = idx >> 2, c = (idx & 3) * 8;
      int grow = rowBase + r; if (grow >= M) grow = M - 1;
      pa0 = *(const float4*)(A + (size_t)grow * K + k0 + c);
    }
    {
      int idx = tid + 256;
      int r = idx >> 2, c = (idx & 3) * 8;
      int grow = rowBase + r; if (grow >= M) grow = M - 1;
      pa1 = *(const float4*)(A + (size_t)grow * K + k0 + c);
    }
    // B tile: 64 rows x 64B -> 256 b128 loads -> 1 per thread
    {
      int r = tid >> 2, c = (tid & 3) * 8;
      pb = *(const float4*)(Wt + (size_t)(colBase + r) * K + k0 + c);
    }
  };
  auto sstore = [&](int b) {
    bf16* a = sA + b * (BM * ASTR);
    bf16* bb = sB + b * (BN * BSTR);
    {
      int idx = tid;
      int r = idx >> 2, c = (idx & 3) * 8;
      *(float4*)(a + r * ASTR + c) = pa0;
    }
    {
      int idx = tid + 256;
      int r = idx >> 2, c = (idx & 3) * 8;
      *(float4*)(a + r * ASTR + c) = pa1;
    }
    {
      int r = tid >> 2, c = (tid & 3) * 8;
      *(float4*)(bb + r * BSTR + c) = pb;
    }
  };

  gload(0);
  sstore(0);
  int buf = 0;

  for (int k0 = 0; k0 < K; k0 += BK) {
    const bool nxt = (k0 + BK) < K;
    __syncthreads();
    if (nxt) gload(k0 + BK);            // global loads for next tile issue early

    const bf16* a = sA + buf * (BM * ASTR);
    const bf16* bbase = sB + buf * (BN * BSTR);

    // A fragment (16x32 bf16): lane<16 -> K 0..15, lane>=16 -> K 16..31
    // halves 0..15 -> K kb..kb+15
    const int r  = lane & 15;
    const int kb = (lane >> 4) * 16;
    const bf16* arow = a + (wave * 16 + r) * ASTR;
    v16bf afrag;
    #pragma unroll
    for (int j = 0; j < 8; ++j)  afrag[j] = arow[kb + j];
    #pragma unroll
    for (int j = 8; j < 16; ++j) afrag[j] = arow[kb + j];

    // B fragments (32x16 bf16): lanes 0-15 hold K 0-15 (col = lane), lanes 16-31 hold K 16-31
    const int kb2 = (lane >> 4) * 16;
    #pragma unroll
    for (int t = 0; t < 4; ++t) {
      const bf16* brow = bbase + (t * 16 + (lane & 15)) * BSTR + kb2;
      v16bf bfrag;
      #pragma unroll
      for (int j = 0; j < 16; ++j) bfrag[j] = brow[j];
      acc[t] = __builtin_amdgcn_wmma_f32_16x16x32_bf16(
          false, afrag, false, bfrag, (short)0, acc[t], false, false);
    }

    if (nxt) sstore(buf ^ 1);           // park next tile after compute issued
    buf ^= 1;
  }

  // C/D layout: VGPR g -> M = g (lanes 0-15) or 8+g (lanes 16-31); N = lane&15
  const int mOff = (lane >> 4) * 8;
  const int nOff = lane & 15;
  #pragma unroll
  for (int t = 0; t < 4; ++t) {
    #pragma unroll
    for (int g = 0; g < 8; ++g) {
      int row = rowBase + wave * 16 + mOff + g;
      int col = colBase + t * 16 + nOff;
      if (row < M) C[(size_t)row * N + col] = acc[t][g];
    }
  }
}

// ---------------- CSR aggregation + self loop + bias + relu + bf16 requant ----------------
// One block per node, 128 threads x float4 features; edge list + norms staged in LDS.

#define AGG_CHUNK 512

__global__ __launch_bounds__(128)
void aggregate(const float4* __restrict__ hW4, const int* __restrict__ row_ptr,
               const int* __restrict__ esrc, const float* __restrict__ dinv,
               const float4* __restrict__ bias4, float4* __restrict__ hout4,
               bf16* __restrict__ aout, int F4 /* = F/4 */) {
  __shared__ int   sIdx[AGG_CHUNK];
  __shared__ float sNrm[AGG_CHUNK];
  const int node = blockIdx.x;
  const int tid  = threadIdx.x;
  const float di = dinv[node];
  const int beg = row_ptr[node], end = row_ptr[node + 1];
  const bool active = tid < F4;
  const size_t rowBase = (size_t)node * F4;

  float4 acc = make_float4(0.f, 0.f, 0.f, 0.f);
  if (active) {                          // self-loop term: norm = di*di
    float4 v = hW4[rowBase + tid];
    float w = di * di;
    acc.x = v.x * w; acc.y = v.y * w; acc.z = v.z * w; acc.w = v.w * w;
  }

  for (int chunk = beg; chunk < end; chunk += AGG_CHUNK) {
    int cnt = end - chunk; if (cnt > AGG_CHUNK) cnt = AGG_CHUNK;
    __syncthreads();
    for (int j = tid; j < cnt; j += 128) {
      int s = esrc[chunk + j];
      sIdx[j] = s;
      sNrm[j] = dinv[s] * di;
    }
    __syncthreads();
    if (active) {
      for (int j = 0; j < cnt; ++j) {
        int s = sIdx[j];
        float w = sNrm[j];
        float4 v = hW4[(size_t)s * F4 + tid];
        acc.x += v.x * w; acc.y += v.y * w; acc.z += v.z * w; acc.w += v.w * w;
      }
    }
  }

  if (active) {
    float4 b = bias4[tid];
    acc.x = fmaxf(acc.x + b.x, 0.f);
    acc.y = fmaxf(acc.y + b.y, 0.f);
    acc.z = fmaxf(acc.z + b.z, 0.f);
    acc.w = fmaxf(acc.w + b.w, 0.f);
    hout4[rowBase + tid] = acc;
    bf16 t4[4] = { (bf16)acc.x, (bf16)acc.y, (bf16)acc.z, (bf16)acc.w };
    *(uint2*)(aout + (rowBase + tid) * 4) = *(uint2*)t4;   // packed b64 store
  }
}

// ---------------- global mean pool ----------------

__global__ void pool_acc(const float* __restrict__ h, const int* __restrict__ batch,
                         float* sums, int* cnt, int F) {
  int node = blockIdx.x;
  int f = threadIdx.x;          // F == 128
  int g = batch[node];
  atomicAdd(&sums[(size_t)g * F + f], h[(size_t)node * F + f]);
  if (f == 0) atomicAdd(&cnt[g], 1);
}
__global__ void pool_fin(const float* __restrict__ sums, const int* __restrict__ cnt,
                         float* __restrict__ out, int F) {
  int g = blockIdx.x;
  int f = threadIdx.x;
  float c = (float)cnt[g];
  if (c < 1.0f) c = 1.0f;
  out[(size_t)g * F + f] = sums[(size_t)g * F + f] / c;
}

// ---------------- launcher ----------------

extern "C" void kernel_launch(void* const* d_in, const int* in_sizes, int n_in,
                              void* d_out, int out_size, void* d_ws, size_t ws_size,
                              hipStream_t stream) {
  (void)in_sizes; (void)n_in; (void)out_size; (void)ws_size;

  const float* x     = (const float*)d_in[0];
  const int*   ei    = (const int*)d_in[1];     // [2][E]: row0 = src, row1 = dst
  const int*   batch = (const int*)d_in[2];
  const float* W[5]; const float* Bv[5];
  for (int i = 0; i < 5; ++i) { W[i] = (const float*)d_in[3 + 2 * i]; Bv[i] = (const float*)d_in[4 + 2 * i]; }
  float* out = (float*)d_out;

  const int fiL[5] = {128, 512, 512, 512, 512};
  const int foL[5] = {512, 512, 512, 512, 128};

  // workspace carve-up
  char* ws = (char*)d_ws;
  size_t off = 0;
  auto alloc = [&](size_t bytes) -> void* {
    off = (off + 255) & ~(size_t)255;
    void* p = ws + off;
    off += bytes;
    return p;
  };
  bf16*  Abuf   = (bf16*)alloc((size_t)N_NODES * 512 * sizeof(bf16));
  float* hWbuf  = (float*)alloc((size_t)N_NODES * 512 * sizeof(float));
  float* hbuf   = (float*)alloc((size_t)N_NODES * 512 * sizeof(float));
  bf16*  Wt[5];
  for (int i = 0; i < 5; ++i) Wt[i] = (bf16*)alloc((size_t)fiL[i] * foL[i] * sizeof(bf16));
  int*   counts = (int*)alloc((size_t)N_NODES * sizeof(int));
  float* dinv   = (float*)alloc((size_t)N_NODES * sizeof(float));
  int*   rowptr = (int*)alloc((size_t)(N_NODES + 1) * sizeof(int));
  int*   cursor = (int*)alloc((size_t)N_NODES * sizeof(int));
  int*   esrc   = (int*)alloc((size_t)N_EDGES * sizeof(int));
  float* psums  = (float*)alloc((size_t)N_GRAPHS * 128 * sizeof(float));
  int*   pcnt   = (int*)alloc((size_t)N_GRAPHS * sizeof(int));

  const int* srcArr = ei;
  const int* dstArr = ei + N_EDGES;

  // --- weight prep (bf16, transposed to [fo][fi]) ---
  for (int i = 0; i < 5; ++i) {
    int n = fiL[i] * foL[i];
    transpose_w<<<(n + 255) / 256, 256, 0, stream>>>(W[i], Wt[i], fiL[i], foL[i]);
  }

  // --- degree / normalization / CSR (edges are identical for all 5 layers) ---
  zero_i32<<<(N_NODES + 255) / 256, 256, 0, stream>>>(counts, N_NODES);
  count_deg<<<(N_EDGES + 255) / 256, 256, 0, stream>>>(dstArr, counts, N_EDGES);
  make_dinv<<<(N_NODES + 255) / 256, 256, 0, stream>>>(counts, dinv, N_NODES);
  scan_rowptr<<<1, 1024, 0, stream>>>(counts, rowptr, N_NODES);
  copy_i32<<<(N_NODES + 255) / 256, 256, 0, stream>>>(rowptr, cursor, N_NODES);
  fill_edges<<<(N_EDGES + 255) / 256, 256, 0, stream>>>(srcArr, dstArr, cursor, esrc, N_EDGES);

  // --- input features to bf16 ---
  {
    int n = N_NODES * 128;
    cvt_f32_bf16<<<(n + 255) / 256, 256, 0, stream>>>(x, Abuf, n);
  }

  // --- 5 GCN layers ---
  for (int l = 0; l < 5; ++l) {
    int K = fiL[l], F = foL[l];
    dim3 ggrid((N_NODES + BM - 1) / BM, F / BN);
    gemm_bf16_wmma<<<ggrid, 256, 0, stream>>>(Abuf, Wt[l], hWbuf, N_NODES, K, F);
    aggregate<<<N_NODES, 128, 0, stream>>>((const float4*)hWbuf, rowptr, esrc, dinv,
                                           (const float4*)Bv[l], (float4*)hbuf, Abuf, F / 4);
  }

  // --- global mean pool over graph ids ---
  zero_f32<<<(N_GRAPHS * 128 + 255) / 256, 256, 0, stream>>>(psums, N_GRAPHS * 128);
  zero_i32<<<1, 64, 0, stream>>>(pcnt, N_GRAPHS);
  pool_acc<<<N_NODES, 128, 0, stream>>>(hbuf, batch, psums, pcnt, 128);
  pool_fin<<<N_GRAPHS, 128, 0, stream>>>(psums, pcnt, out, 128);
}
